// RumourGAT_39178691674466
// MI455X (gfx1250) — matrix operand order repaired and checked by reading, not compiled, not run on hardware
//
#include <hip/hip_runtime.h>
#include <hip/hip_bf16.h>
#include <cfloat>

typedef __attribute__((ext_vector_type(16))) _Float16 v16h;
typedef __attribute__((ext_vector_type(8)))  _Float16 v8h;
typedef __attribute__((ext_vector_type(8)))  float    v8f;

#define BN_EPS     1e-5f
#define NEG_SLOPE  0.2f

// ---------- order-preserving float <-> uint encoding for atomicMax ----------
__device__ __forceinline__ unsigned encF(float f) {
  unsigned u = __float_as_uint(f);
  return (u & 0x80000000u) ? ~u : (u | 0x80000000u);
}
__device__ __forceinline__ float decF(unsigned e) {
  unsigned u = (e & 0x80000000u) ? (e & 0x7fffffffu) : ~e;
  return __uint_as_float(u);
}

// ---------------------------- utility kernels -------------------------------
__global__ void k_f32_to_f16(const float* __restrict__ src, _Float16* __restrict__ dst, long n) {
  long t = (long)blockIdx.x * blockDim.x + threadIdx.x;
  if (t < n) dst[t] = (_Float16)src[t];
}

__global__ void k_zero_f32(float* __restrict__ p, long n) {
  long t = (long)blockIdx.x * blockDim.x + threadIdx.x;
  if (t < n) p[t] = 0.0f;
}

__global__ void k_init_seg(unsigned* __restrict__ seg, float* __restrict__ den, long n) {
  long t = (long)blockIdx.x * blockDim.x + threadIdx.x;
  if (t < n) { seg[t] = encF(-FLT_MAX); den[t] = 0.0f; }
}

// Pack row-major f32 weight W[K x Ncols] into WMMA B-fragment order (f16):
// Bp[(((kt*NT)+nt)*32 + lane)*16 + i] = W[kt*32 + (lane<16?0:16) + i][nt*16 + (lane&15)]
__global__ void k_pack_B(const float* __restrict__ W, _Float16* __restrict__ Bp,
                         int K, int Ncols) {
  int total = K * Ncols;
  int t = blockIdx.x * blockDim.x + threadIdx.x;
  if (t >= total) return;
  int i    = t & 15;
  int lane = (t >> 4) & 31;
  int rest = t >> 9;                 // (kt*NT + nt)
  int NT   = Ncols >> 4;
  int nt   = rest % NT;
  int kt   = rest / NT;
  int k = kt * 32 + ((lane < 16) ? 0 : 16) + i;
  int n = nt * 16 + (lane & 15);
  Bp[t] = (_Float16)W[(size_t)k * Ncols + n];
}

// ------------------------------ WMMA GEMM -----------------------------------
// C[Mrows x N] = A[Mrows x K] * B[K x N], A f16 row-major, B pre-packed f16,
// f32 accumulate. One wave per 16-row strip; all A K-fragments kept in regs.
// Per N-tile: preload all KT B-fragments (clause of b128 loads, single wait),
// then KT back-to-back WMMAs; stores take a wave-uniform full-tile fast path.
template<int K, int N>
__global__ void k_wmma_gemm(const _Float16* __restrict__ A,
                            const _Float16* __restrict__ Bp,
                            float* __restrict__ C, int Mrows) {
  constexpr int KT = K / 32;
  constexpr int NT = N / 16;
  const int lane = threadIdx.x & 31;
  const int tileM = blockIdx.x * (blockDim.x >> 5) + (threadIdx.x >> 5);
  const int Mtiles = (Mrows + 15) >> 4;
  if (tileM >= Mtiles) return;

  int rowA = tileM * 16 + (lane & 15);
  if (rowA >= Mrows) rowA = Mrows - 1;          // clamp (EXEC stays all-ones)
  const int kb = (lane < 16) ? 0 : 16;          // K sub-block per lane half

  v16h a[KT];
  const _Float16* Ar = A + (size_t)rowA * K;
#pragma unroll
  for (int kt = 0; kt < KT; ++kt) {
    // lane<16: K={0..15}; lane>=16: K={16..31} (contiguous)
    ((v8h*)&a[kt])[0] = *(const v8h*)(Ar + kt * 32 + kb);
    ((v8h*)&a[kt])[1] = *(const v8h*)(Ar + kt * 32 + kb + 8);
  }

  const int rowD = tileM * 16 + ((lane < 16) ? 0 : 8);
  const bool fullTile = (tileM * 16 + 16) <= Mrows;    // wave-uniform
  const _Float16* BpL = Bp + (size_t)lane * 16;

  for (int nt = 0; nt < NT; ++nt) {
    // preload all B fragments for this N-tile
    v16h b[KT];
#pragma unroll
    for (int kt = 0; kt < KT; ++kt) {
      const _Float16* bp = BpL + (size_t)(kt * NT + nt) * 512;
      ((v8h*)&b[kt])[0] = *(const v8h*)bp;
      ((v8h*)&b[kt])[1] = *(const v8h*)(bp + 8);
    }
    v8f acc = {0.f, 0.f, 0.f, 0.f, 0.f, 0.f, 0.f, 0.f};
#pragma unroll
    for (int kt = 0; kt < KT; ++kt) {
      acc = __builtin_amdgcn_wmma_f32_16x16x32_f16(
          /*neg_a=*/false, a[kt], /*neg_b=*/false, b[kt],
          /*c_mod=*/(short)0, acc, /*reuse_a=*/false, /*reuse_b=*/false);
    }
    const int col = nt * 16 + (lane & 15);
    float* Cp = C + (size_t)rowD * N + col;
    if (fullTile) {
#pragma unroll
      for (int r = 0; r < 8; ++r) Cp[(size_t)r * N] = acc[r];
    } else {
#pragma unroll
      for (int r = 0; r < 8; ++r)
        if (rowD + r < Mrows) Cp[(size_t)r * N] = acc[r];
    }
  }
}

// --------------------- attention logits: a = <h[n,h,:], att[h,:]> -----------
template<int H, int C>
__global__ void k_att_dot(const float* __restrict__ Hm, const float* __restrict__ attS,
                          const float* __restrict__ attD, float* __restrict__ aS,
                          float* __restrict__ aD, int N) {
  constexpr int FP  = H * C / 32;   // floats per lane
  constexpr int LPH = 32 / H;       // lanes per head
  int node = blockIdx.x * (blockDim.x >> 5) + (threadIdx.x >> 5);
  if (node >= N) return;
  int lane = threadIdx.x & 31;
  const float* row = Hm + (size_t)node * H * C + lane * FP;
  const float* as  = attS + lane * FP;       // att is [H,C] flat => same index
  const float* ad  = attD + lane * FP;
  float s = 0.f, d = 0.f;
#pragma unroll
  for (int i = 0; i < FP; ++i) { float v = row[i]; s += v * as[i]; d += v * ad[i]; }
#pragma unroll
  for (int m = 1; m < LPH; m <<= 1) { s += __shfl_xor(s, m, 32); d += __shfl_xor(d, m, 32); }
  if ((lane & (LPH - 1)) == 0) {
    int h = lane / LPH;
    aS[(size_t)node * H + h] = s;
    aD[(size_t)node * H + h] = d;
  }
}

// -------------------------- edge softmax passes -----------------------------
__global__ void k_edge_max(const int* __restrict__ esrc, const int* __restrict__ edst,
                           int E, int N, const float* __restrict__ aS,
                           const float* __restrict__ aD, unsigned* __restrict__ seg, int H) {
  long Etot = (long)E + N;
  long e = (long)blockIdx.x * blockDim.x + threadIdx.x;
  if (e >= Etot) return;
  int s = (e < E) ? esrc[e] : (int)(e - E);
  int d = (e < E) ? edst[e] : (int)(e - E);
  for (int h = 0; h < H; ++h) {
    float v = aS[(size_t)s * H + h] + aD[(size_t)d * H + h];
    v = (v > 0.f) ? v : NEG_SLOPE * v;
    atomicMax(&seg[(size_t)d * H + h], encF(v));
  }
}

__global__ void k_edge_exp(const int* __restrict__ esrc, const int* __restrict__ edst,
                           int E, int N, const float* __restrict__ aS,
                           const float* __restrict__ aD, const unsigned* __restrict__ seg,
                           float* __restrict__ ex, float* __restrict__ den, int H) {
  long Etot = (long)E + N;
  long e = (long)blockIdx.x * blockDim.x + threadIdx.x;
  if (e >= Etot) return;
  int s = (e < E) ? esrc[e] : (int)(e - E);
  int d = (e < E) ? edst[e] : (int)(e - E);
  for (int h = 0; h < H; ++h) {
    float v = aS[(size_t)s * H + h] + aD[(size_t)d * H + h];
    v = (v > 0.f) ? v : NEG_SLOPE * v;
    float m = decF(seg[(size_t)d * H + h]);
    float xv = __expf(v - m);
    ex[(size_t)e * H + h] = xv;
    atomicAdd(&den[(size_t)d * H + h], xv);
  }
}

// wave-per-edge weighted scatter: out[dst] += h[src] * alpha
template<int H, int C>
__global__ void k_edge_aggr(const int* __restrict__ esrc, const int* __restrict__ edst,
                            int E, int N, const float* __restrict__ Hm,
                            const float* __restrict__ ex, const float* __restrict__ den,
                            float* __restrict__ out) {
  constexpr int FP = H * C / 32;
  long Etot = (long)E + N;
  long e = (long)blockIdx.x * (blockDim.x >> 5) + (threadIdx.x >> 5);
  if (e >= Etot) return;
  int lane = threadIdx.x & 31;
  int s = (e < E) ? esrc[e] : (int)(e - E);
  int d = (e < E) ? edst[e] : (int)(e - E);
  int j0 = lane * FP;
  int h = j0 / C;
  float alpha = ex[(size_t)e * H + h] / (den[(size_t)d * H + h] + 1e-16f);
  const float* hs = Hm + (size_t)s * H * C + j0;
  float* od = out + (size_t)d * H * C + j0;
#pragma unroll
  for (int i = 0; i < FP; ++i) atomicAdd(&od[i], hs[i] * alpha);
}

// --------------------------- BN + ELU (+bias) -------------------------------
__global__ void k_bn_elu_f32(const float* __restrict__ x, const float* __restrict__ bias,
                             const float* __restrict__ g, const float* __restrict__ b,
                             const float* __restrict__ m, const float* __restrict__ v,
                             float* __restrict__ y, long n, int C) {
  long t = (long)blockIdx.x * blockDim.x + threadIdx.x;
  if (t >= n) return;
  int c = (int)(t % (long)C);
  float val = (x[t] + bias[c] - m[c]) * rsqrtf(v[c] + BN_EPS) * g[c] + b[c];
  y[t] = (val > 0.f) ? val : (__expf(val) - 1.f);
}

__global__ void k_bn_elu_f16(const float* __restrict__ x, const float* __restrict__ bias,
                             const float* __restrict__ g, const float* __restrict__ b,
                             const float* __restrict__ m, const float* __restrict__ v,
                             _Float16* __restrict__ y, long n, int C) {
  long t = (long)blockIdx.x * blockDim.x + threadIdx.x;
  if (t >= n) return;
  int c = (int)(t % (long)C);
  float val = (x[t] + bias[c] - m[c]) * rsqrtf(v[c] + BN_EPS) * g[c] + b[c];
  val = (val > 0.f) ? val : (__expf(val) - 1.f);
  y[t] = (_Float16)val;
}

// ------------------------------- readout ------------------------------------
__global__ void k_readout_init(float* __restrict__ meanA, unsigned* __restrict__ maxA) {
  int t = threadIdx.x;
  if (t < 64) { meanA[t] = 0.f; maxA[t] = encF(-FLT_MAX); }
}

__global__ void k_readout(const float* __restrict__ emb, float* __restrict__ meanA,
                          unsigned* __restrict__ maxA, int N) {
  __shared__ float ss[256];
  __shared__ float sm[256];
  int c   = threadIdx.x & 63;
  int grp = threadIdx.x >> 6;
  float s = 0.f, m = -FLT_MAX;
  for (int r = blockIdx.x * 4 + grp; r < N; r += gridDim.x * 4) {
    float v = emb[(size_t)r * 64 + c];
    s += v;
    m = fmaxf(m, v);
  }
  ss[threadIdx.x] = s;
  sm[threadIdx.x] = m;
  __syncthreads();
  if (threadIdx.x < 64) {
    float s2 = ss[c] + ss[c + 64] + ss[c + 128] + ss[c + 192];
    float m2 = fmaxf(fmaxf(sm[c], sm[c + 64]), fmaxf(sm[c + 128], sm[c + 192]));
    atomicAdd(&meanA[c], s2);
    atomicMax(&maxA[c], encF(m2));
  }
}

// ------------------------------ final MLP -----------------------------------
__global__ void k_final(const float* __restrict__ meanA, const unsigned* __restrict__ maxA,
                        const float* __restrict__ emb, const int* __restrict__ rootp, int N,
                        const float* __restrict__ w1, const float* __restrict__ b1_,
                        const float* __restrict__ cg, const float* __restrict__ cb,
                        const float* __restrict__ cm, const float* __restrict__ cv,
                        const float* __restrict__ w2, const float* __restrict__ b2_,
                        const float* __restrict__ w3, const float* __restrict__ b3_,
                        float* __restrict__ out) {
  __shared__ float gf[192];
  __shared__ float s1[64];
  __shared__ float s2[32];
  int t = threadIdx.x;
  int root = rootp[0];
  if (root > N - 1) root = N - 1;
  if (root < 0) root = 0;
  gf[t]        = meanA[t] * (1.0f / (float)N);
  gf[64 + t]   = decF(maxA[t]);
  gf[128 + t]  = emb[(size_t)root * 64 + t];
  __syncthreads();
  float acc = b1_[t];
  for (int k = 0; k < 192; ++k) acc += gf[k] * w1[k * 64 + t];
  acc = (acc - cm[t]) * rsqrtf(cv[t] + BN_EPS) * cg[t] + cb[t];
  s1[t] = fmaxf(acc, 0.f);
  __syncthreads();
  if (t < 32) {
    float a2 = b2_[t];
    for (int k = 0; k < 64; ++k) a2 += s1[k] * w2[k * 32 + t];
    s2[t] = fmaxf(a2, 0.f);
  }
  __syncthreads();
  if (t == 0) {
    float a3 = b3_[0];
    for (int k = 0; k < 32; ++k) a3 += s2[k] * w3[k];
    out[0] = a3;
  }
}

// ---------------------------------------------------------------------------
extern "C" void kernel_launch(void* const* d_in, const int* in_sizes, int n_in,
                              void* d_out, int out_size, void* d_ws, size_t ws_size,
                              hipStream_t stream) {
  const float* x     = (const float*)d_in[0];
  const int*   eidx  = (const int*)d_in[1];
  const int*   rootp = (const int*)d_in[2];
  const float* W1    = (const float*)d_in[3];
  const float* attS1 = (const float*)d_in[4];
  const float* attD1 = (const float*)d_in[5];
  const float* b1    = (const float*)d_in[6];
  const float* W2    = (const float*)d_in[7];
  const float* attS2 = (const float*)d_in[8];
  const float* attD2 = (const float*)d_in[9];
  const float* b2    = (const float*)d_in[10];
  const float* bn1g  = (const float*)d_in[11];
  const float* bn1b  = (const float*)d_in[12];
  const float* bn1m  = (const float*)d_in[13];
  const float* bn1v  = (const float*)d_in[14];
  const float* bn2g  = (const float*)d_in[15];
  const float* bn2b  = (const float*)d_in[16];
  const float* bn2m  = (const float*)d_in[17];
  const float* bn2v  = (const float*)d_in[18];
  const float* cg1w  = (const float*)d_in[19];
  const float* cg1b  = (const float*)d_in[20];
  const float* cbng  = (const float*)d_in[21];
  const float* cbnb  = (const float*)d_in[22];
  const float* cbnm  = (const float*)d_in[23];
  const float* cbnv  = (const float*)d_in[24];
  const float* cg2w  = (const float*)d_in[25];
  const float* cg2b  = (const float*)d_in[26];
  const float* cg3w  = (const float*)d_in[27];
  const float* cg3b  = (const float*)d_in[28];

  const int N = in_sizes[0] / 128;   // 100000
  const int E = in_sizes[1] / 2;     // 1600000
  const int Etot = E + N;
  const int* esrc = eidx;
  const int* edst = eidx + E;

  // ------------------------- workspace layout -------------------------------
  char* base = (char*)d_ws;
  size_t off = 0;
  auto carve = [&](size_t bytes) -> char* {
    char* p = base + off;
    off = (off + bytes + 255) & ~(size_t)255;
    return p;
  };
  float*    Hbuf  = (float*)carve((size_t)N * 256 * 4);     // H1, later H2
  float*    OUT   = (float*)carve((size_t)N * 256 * 4);     // out1, later out2/emb
  _Float16* F16A  = (_Float16*)carve((size_t)N * 256 * 2);  // x-f16, later h2-f16
  _Float16* BP    = (_Float16*)carve((size_t)128 * 256 * 2);// packed weights
  float*    AS    = (float*)carve((size_t)N * 4 * 4);
  float*    AD    = (float*)carve((size_t)N * 4 * 4);
  unsigned* SEG   = (unsigned*)carve((size_t)N * 4 * 4);
  float*    DEN   = (float*)carve((size_t)N * 4 * 4);
  float*    EXB   = (float*)carve((size_t)Etot * 4 * 4);
  float*    MEANA = (float*)carve(64 * 4);
  unsigned* MAXA  = (unsigned*)carve(64 * 4);

  const int TB = 256;
  const int Mtiles = (N + 15) / 16;
  const int gemmGrid = (Mtiles + 7) / 8;

  // ------------------------------ layer 1 -----------------------------------
  { long n = (long)N * 128;
    k_f32_to_f16<<<(unsigned)((n + TB - 1) / TB), TB, 0, stream>>>(x, F16A, n); }
  k_pack_B<<<(128 * 256 + TB - 1) / TB, TB, 0, stream>>>(W1, BP, 128, 256);
  k_wmma_gemm<128, 256><<<gemmGrid, TB, 0, stream>>>(F16A, BP, Hbuf, N);

  k_att_dot<4, 64><<<(N + 7) / 8, TB, 0, stream>>>(Hbuf, attS1, attD1, AS, AD, N);

  { long n = (long)N * 4;
    k_init_seg<<<(unsigned)((n + TB - 1) / TB), TB, 0, stream>>>(SEG, DEN, n); }
  { long n = (long)N * 256;
    k_zero_f32<<<(unsigned)((n + TB - 1) / TB), TB, 0, stream>>>(OUT, n); }

  k_edge_max<<<(Etot + TB - 1) / TB, TB, 0, stream>>>(esrc, edst, E, N, AS, AD, SEG, 4);
  k_edge_exp<<<(Etot + TB - 1) / TB, TB, 0, stream>>>(esrc, edst, E, N, AS, AD, SEG, EXB, DEN, 4);
  k_edge_aggr<4, 64><<<(Etot + 7) / 8, TB, 0, stream>>>(esrc, edst, E, N, Hbuf, EXB, DEN, OUT);

  // BN1 + ELU (+b1), output f16 feeding GEMM2
  { long n = (long)N * 256;
    k_bn_elu_f16<<<(unsigned)((n + TB - 1) / TB), TB, 0, stream>>>(
        OUT, b1, bn1g, bn1b, bn1m, bn1v, F16A, n, 256); }

  // ------------------------------ layer 2 -----------------------------------
  k_pack_B<<<(256 * 64 + TB - 1) / TB, TB, 0, stream>>>(W2, BP, 256, 64);
  k_wmma_gemm<256, 64><<<gemmGrid, TB, 0, stream>>>(F16A, BP, Hbuf, N);

  k_att_dot<1, 64><<<(N + 7) / 8, TB, 0, stream>>>(Hbuf, attS2, attD2, AS, AD, N);

  { long n = (long)N;
    k_init_seg<<<(unsigned)((n + TB - 1) / TB), TB, 0, stream>>>(SEG, DEN, n); }
  { long n = (long)N * 64;
    k_zero_f32<<<(unsigned)((n + TB - 1) / TB), TB, 0, stream>>>(OUT, n); }

  k_edge_max<<<(Etot + TB - 1) / TB, TB, 0, stream>>>(esrc, edst, E, N, AS, AD, SEG, 1);
  k_edge_exp<<<(Etot + TB - 1) / TB, TB, 0, stream>>>(esrc, edst, E, N, AS, AD, SEG, EXB, DEN, 1);
  k_edge_aggr<1, 64><<<(Etot + 7) / 8, TB, 0, stream>>>(esrc, edst, E, N, Hbuf, EXB, DEN, OUT);

  // BN2 + ELU (+b2), in-place -> node embeddings [N,64]
  { long n = (long)N * 64;
    k_bn_elu_f32<<<(unsigned)((n + TB - 1) / TB), TB, 0, stream>>>(
        OUT, b2, bn2g, bn2b, bn2m, bn2v, OUT, n, 64); }

  // ------------------------------ readout + MLP -----------------------------
  k_readout_init<<<1, 64, 0, stream>>>(MEANA, MAXA);
  k_readout<<<512, TB, 0, stream>>>(OUT, MEANA, MAXA, N);
  k_final<<<1, 64, 0, stream>>>(MEANA, MAXA, OUT, rootp, N,
                                cg1w, cg1b, cbng, cbnb, cbnm, cbnv,
                                cg2w, cg2b, cg3w, cg3b, (float*)d_out);
}
